// BidirectionalMamba_41618233098840
// MI455X (gfx1250) — matrix operand order repaired and checked by reading, not compiled, not run on hardware
//
#include <hip/hip_runtime.h>
#include <stdint.h>

#define D_MODELc 512
#define D_INNERc 1024
#define D_STATEc 16
#define DT_RANKc 32
#define BATCHc   4
#define SEQc     2048
#define MROWSc   (BATCHc * SEQc)     // 8192
#define NCHUNKc  8
#define CHUNKc   (SEQc / NCHUNKc)    // 256

typedef __attribute__((ext_vector_type(16))) __bf16 bf16x16;
typedef __attribute__((ext_vector_type(8)))  __bf16 bf16x8;
typedef __attribute__((ext_vector_type(4)))  __bf16 bf16x4;
typedef __attribute__((ext_vector_type(8)))  float  f32x8;
typedef __attribute__((ext_vector_type(4)))  unsigned int u32x4;
typedef __attribute__((ext_vector_type(8)))  int i32x8;
typedef __attribute__((ext_vector_type(4)))  int i32x4;

#if defined(__HIP_DEVICE_COMPILE__) && __has_builtin(__builtin_amdgcn_tensor_load_to_lds)
#define HAVE_TDM 1
#else
#define HAVE_TDM 0
#endif

__device__ __forceinline__ float silu_f(float v) { return v / (1.f + __expf(-v)); }
__device__ __forceinline__ float softplus_f(float v) { return v > 20.f ? v : log1pf(__expf(v)); }

#if HAVE_TDM
// ---------------------------------------------------------------------------
// TDM: issue a 2D bf16 tile load (tile_d0 x tile_d1, dim0 contiguous) into LDS.
// Descriptor packed per CDNA5 ISA Tensor-DMA D# spec (CDNA5 ISA).
// 6-arg builtin form (clang-23 / therock headers): extension groups zeroed.
// ---------------------------------------------------------------------------
__device__ __forceinline__ void tdm_load_2d(uint32_t lds_off, uint64_t gaddr,
                                            uint32_t tdim0, uint32_t tdim1,
                                            uint64_t stride0,
                                            uint32_t tile_d0, uint32_t tile_d1) {
  u32x4 g0;
  g0[0] = 1u;                                   // count=1, user descriptor
  g0[1] = lds_off;                              // LDS byte address
  g0[2] = (uint32_t)gaddr;                      // global addr [31:0]
  g0[3] = (uint32_t)((gaddr >> 32) & 0x01FFFFFFu) | (2u << 30); // [56:32] | type=2
  i32x8 g1;
  g1[0] = (int)(1u << 16);                      // data_size=1 (2B), mask=0
  g1[1] = (int)((tdim0 & 0xFFFFu) << 16);       // atomic_addr=0 | tensor_dim0 lo
  g1[2] = (int)((tdim0 >> 16) | ((tdim1 & 0xFFFFu) << 16));
  g1[3] = (int)((tdim1 >> 16) | (tile_d0 << 16));
  g1[4] = (int)(tile_d1 & 0xFFFFu);             // tile_dim1 | tile_dim2=0
  g1[5] = (int)(uint32_t)(stride0 & 0xFFFFFFFFu);
  g1[6] = (int)(uint32_t)((stride0 >> 32) & 0xFFFFu); // | dim1_stride lo = 0
  g1[7] = 0;
  i32x4 gz4 = {0, 0, 0, 0};
  i32x8 gz8 = {0, 0, 0, 0, 0, 0, 0, 0};
  __builtin_amdgcn_tensor_load_to_lds(g0, g1, gz4, gz4, gz8, 0);
}
#endif

// ---------------------------------------------------------------------------
// A-fragment: 16x32 bf16 (MxK), ISA 7.12.2 layout:
//   lanes 0-15 : row=lane,  K = kb+0..7  and kb+16..23
//   lanes 16-31: row=l-16,  K = kb+8..15 and kb+24..31
// ---------------------------------------------------------------------------
template<bool FLIP>
__device__ __forceinline__ bf16x16 load_a_frag(const __bf16* __restrict__ A, int lda,
                                               int r0, int kb, int lane) {
  int lr = lane & 15, hi = lane >> 4;
  int r = r0 + lr;
  if (FLIP) { int b = r >> 11; int t = r & (SEQc - 1); r = (b << 11) + (SEQc - 1 - t); }
  const __bf16* p = A + (size_t)r * lda + kb + hi * 8;
  bf16x8 lo = *(const bf16x8*)(p);
  bf16x8 hh = *(const bf16x8*)(p + 16);
  return __builtin_shufflevector(lo, hh, 0, 1, 2, 3, 4, 5, 6, 7,
                                         8, 9, 10, 11, 12, 13, 14, 15);
}

// ---------------------------------------------------------------------------
// GEMM (NT): C[M,N] = A[M,K] * Bw[N,K]^T, bf16 in / WMMA bf16 / f32 acc.
// Wave tile 64x64 (4x4 frags). B tile staged in LDS via TDM (double-buffered).
// EPI: 0/2 = store, 1 = softplus(v+bias[col]), 3 = flip-add, 4 = f32 + bf16 aux.
// ---------------------------------------------------------------------------
template<int BWM, int BWN, int EPI, bool FLIPA>
__global__ __launch_bounds__(BWM * BWN * 32)
void gemm_nt_bf16(const __bf16* __restrict__ A, int lda,
                  const __bf16* __restrict__ Bw, int ldb,
                  float* __restrict__ C, int ldc,
                  int K, int N,
                  const float* __restrict__ bias,
                  __bf16* __restrict__ auxbf) {
  constexpr int BN = BWN * 64;
  constexpr int NTHR = BWM * BWN * 32;
  __shared__ __bf16 Bsh[2][BN * 32];
  int lane = threadIdx.x & 31;
  int wave = threadIdx.x >> 5;
  int wm = wave % BWM, wn = wave / BWM;
  int mrow0 = (blockIdx.y * BWM + wm) * 64;
  int ncol0blk = blockIdx.x * BN;
  int lr = lane & 15, hi = lane >> 4;

  f32x8 acc[4][4];
#pragma unroll
  for (int mi = 0; mi < 4; ++mi)
#pragma unroll
    for (int ni = 0; ni < 4; ++ni)
#pragma unroll
      for (int j = 0; j < 8; ++j) acc[mi][ni][j] = 0.f;

  const int nk = K / 32;
#if HAVE_TDM
  if (wave == 0)
    tdm_load_2d((uint32_t)(uintptr_t)&Bsh[0][0],
                (uint64_t)(uintptr_t)(Bw + (size_t)ncol0blk * ldb),
                (uint32_t)K, (uint32_t)N, (uint64_t)ldb, 32u, (uint32_t)BN);
#endif
  for (int kt = 0; kt < nk; ++kt) {
    int cur = kt & 1;
#if HAVE_TDM
    if (wave == 0) {
      if (kt + 1 < nk) {
        tdm_load_2d((uint32_t)(uintptr_t)&Bsh[cur ^ 1][0],
                    (uint64_t)(uintptr_t)(Bw + (size_t)ncol0blk * ldb + (kt + 1) * 32),
                    (uint32_t)K, (uint32_t)N, (uint64_t)ldb, 32u, (uint32_t)BN);
        __builtin_amdgcn_s_wait_tensorcnt(1);
      } else {
        __builtin_amdgcn_s_wait_tensorcnt(0);
      }
    }
    __syncthreads();
#else
    // fallback: cooperative global->LDS copy of tile kt
    for (int i = threadIdx.x; i < BN * 4; i += NTHR) {
      int rown = i >> 2, cc = (i & 3) * 8;
      *(bf16x8*)&Bsh[cur][rown * 32 + cc] =
          *(const bf16x8*)(Bw + (size_t)(ncol0blk + rown) * ldb + kt * 32 + cc);
    }
    __syncthreads();
#endif
    int kb = kt * 32;
    bf16x16 af[4], bfr[4];
#pragma unroll
    for (int mi = 0; mi < 4; ++mi)
      af[mi] = load_a_frag<FLIPA>(A, lda, mrow0 + mi * 16, kb, lane);
#pragma unroll
    for (int ni = 0; ni < 4; ++ni) {
      const __bf16* bp = &Bsh[cur][(wn * 64 + ni * 16 + lr) * 32 + hi * 8];
      bf16x8 blo = *(const bf16x8*)(bp);
      bf16x8 bhh = *(const bf16x8*)(bp + 16);
      bfr[ni] = __builtin_shufflevector(blo, bhh, 0, 1, 2, 3, 4, 5, 6, 7,
                                        8, 9, 10, 11, 12, 13, 14, 15);
    }
#pragma unroll
    for (int mi = 0; mi < 4; ++mi)
#pragma unroll
      for (int ni = 0; ni < 4; ++ni)
        acc[mi][ni] = __builtin_amdgcn_wmma_f32_16x16x32_bf16(
            false, af[mi], false, bfr[ni], (short)0, acc[mi][ni], false, false);
    __syncthreads();
  }

#pragma unroll
  for (int mi = 0; mi < 4; ++mi)
#pragma unroll
    for (int ni = 0; ni < 4; ++ni)
#pragma unroll
      for (int j = 0; j < 8; ++j) {
        int row = mrow0 + mi * 16 + hi * 8 + j;
        int col = ncol0blk + wn * 64 + ni * 16 + lr;
        float v = acc[mi][ni][j];
        if (EPI == 1) {
          C[(size_t)row * ldc + col] = softplus_f(v + bias[col]);
        } else if (EPI == 3) {
          int b = row >> 11; int t = row & (SEQc - 1);
          size_t o = (size_t)((b << 11) + (SEQc - 1 - t)) * ldc + col;
          C[o] += v;
        } else if (EPI == 4) {
          C[(size_t)row * ldc + col] = v;
          auxbf[(size_t)row * ldc + col] = (__bf16)v;
        } else {
          C[(size_t)row * ldc + col] = v;
        }
      }
}

// ---------------------------------------------------------------------------
// f32 -> bf16 pack (vectorized x4)
// ---------------------------------------------------------------------------
__global__ void pack_bf16_kernel(const float* __restrict__ src,
                                 __bf16* __restrict__ dst, int n4) {
  int i = blockIdx.x * blockDim.x + threadIdx.x;
  if (i < n4) {
    float4 v = ((const float4*)src)[i];
    bf16x4 o;
    o[0] = (__bf16)v.x; o[1] = (__bf16)v.y; o[2] = (__bf16)v.z; o[3] = (__bf16)v.w;
    ((bf16x4*)dst)[i] = o;
  }
}

// ---------------------------------------------------------------------------
// Depthwise causal conv (width 4) + bias + SiLU. Dual store: f32 + bf16.
// ---------------------------------------------------------------------------
__global__ void conv_silu_kernel(const float* __restrict__ xz,
                                 const float* __restrict__ w,
                                 const float* __restrict__ bias,
                                 float* __restrict__ xc,
                                 __bf16* __restrict__ xcb) {
  int idx = blockIdx.x * blockDim.x + threadIdx.x;   // M * 1024
  int c = idx & (D_INNERc - 1);
  int m = idx >> 10;
  int t = m & (SEQc - 1);
  float acc = bias[c];
#pragma unroll
  for (int j = 0; j < 4; ++j) {
    int tt = t - 3 + j;
    if (tt >= 0)
      acc = fmaf(xz[(size_t)(m - 3 + j) * (2 * D_INNERc) + c], w[c * 4 + j], acc);
  }
  float r = silu_f(acc);
  xc[(size_t)idx] = r;
  xcb[(size_t)idx] = (__bf16)r;
}

// ---------------------------------------------------------------------------
// Scan pass A: per chunk propagator P = prod(dA), zero-state response S.
// ---------------------------------------------------------------------------
__global__ __launch_bounds__(256) void scan_reduce_kernel(
    const float* __restrict__ dt, const float* __restrict__ xc,
    const float* __restrict__ xdbl, const float* __restrict__ A_log,
    float* __restrict__ P, float* __restrict__ S) {
  __shared__ float Bsh[CHUNKc][D_STATEc];
  int tid = threadIdx.x;
  int c = blockIdx.x * 256 + tid;
  int ch = blockIdx.y, b = blockIdx.z;
  size_t rowbase = (size_t)b * SEQc + ch * CHUNKc;
  for (int i = tid; i < CHUNKc * D_STATEc; i += 256) {
    int t = i >> 4, s = i & 15;
    Bsh[t][s] = xdbl[(rowbase + t) * 64 + DT_RANKc + s];
  }
  __syncthreads();
  float Av[D_STATEc], Pr[D_STATEc], Sr[D_STATEc];
#pragma unroll
  for (int s = 0; s < D_STATEc; ++s) {
    Av[s] = -__expf(A_log[c * D_STATEc + s]);
    Pr[s] = 1.f; Sr[s] = 0.f;
  }
  for (int t = 0; t < CHUNKc; ++t) {
    float dtv = dt[(rowbase + t) * D_INNERc + c];
    float xv  = xc[(rowbase + t) * D_INNERc + c];
    float du = dtv * xv;
#pragma unroll
    for (int s = 0; s < D_STATEc; ++s) {
      float dA = __expf(dtv * Av[s]);
      Sr[s] = fmaf(dA, Sr[s], du * Bsh[t][s]);
      Pr[s] *= dA;
    }
  }
  size_t ob = (((size_t)(b * NCHUNKc + ch) * D_INNERc) + c) * D_STATEc;
#pragma unroll
  for (int s = 0; s < D_STATEc; ++s) { P[ob + s] = Pr[s]; S[ob + s] = Sr[s]; }
}

// ---------------------------------------------------------------------------
// Scan pass B: combine 8 chunk summaries sequentially -> per-chunk h0.
// ---------------------------------------------------------------------------
__global__ void scan_combine_kernel(const float* __restrict__ P,
                                    const float* __restrict__ S,
                                    float* __restrict__ HI) {
  int idx = blockIdx.x * blockDim.x + threadIdx.x;  // B*1024*16
  int s = idx & 15;
  int c = (idx >> 4) & (D_INNERc - 1);
  int b = idx >> 14;
  float h = 0.f;
#pragma unroll
  for (int k = 0; k < NCHUNKc; ++k) {
    size_t o = (((size_t)(b * NCHUNKc + k) * D_INNERc) + c) * D_STATEc + s;
    HI[o] = h;
    h = fmaf(P[o], h, S[o]);
  }
}

// ---------------------------------------------------------------------------
// Scan pass C: replay chunk with correct h0, produce y, fuse
// (y + xi*D) * silu(z). Output stored directly as bf16 for GEMM4.
// ---------------------------------------------------------------------------
__global__ __launch_bounds__(256) void scan_apply_kernel(
    const float* __restrict__ dt, const float* __restrict__ xc,
    const float* __restrict__ xdbl, const float* __restrict__ A_log,
    const float* __restrict__ HI, const float* __restrict__ xz,
    const float* __restrict__ Dp, __bf16* __restrict__ YFb) {
  __shared__ float Bsh[CHUNKc][D_STATEc];
  __shared__ float Csh[CHUNKc][D_STATEc];
  int tid = threadIdx.x;
  int c = blockIdx.x * 256 + tid;
  int ch = blockIdx.y, b = blockIdx.z;
  size_t rowbase = (size_t)b * SEQc + ch * CHUNKc;
  for (int i = tid; i < CHUNKc * D_STATEc; i += 256) {
    int t = i >> 4, s = i & 15;
    Bsh[t][s] = xdbl[(rowbase + t) * 64 + DT_RANKc + s];
    Csh[t][s] = xdbl[(rowbase + t) * 64 + DT_RANKc + D_STATEc + s];
  }
  __syncthreads();
  float Av[D_STATEc], h[D_STATEc];
  size_t hb = (((size_t)(b * NCHUNKc + ch) * D_INNERc) + c) * D_STATEc;
#pragma unroll
  for (int s = 0; s < D_STATEc; ++s) {
    Av[s] = -__expf(A_log[c * D_STATEc + s]);
    h[s] = HI[hb + s];
  }
  float dcoef = Dp[c];
  for (int t = 0; t < CHUNKc; ++t) {
    float dtv = dt[(rowbase + t) * D_INNERc + c];
    float xv  = xc[(rowbase + t) * D_INNERc + c];
    float du = dtv * xv;
    float y = 0.f;
#pragma unroll
    for (int s = 0; s < D_STATEc; ++s) {
      float dA = __expf(dtv * Av[s]);
      h[s] = fmaf(dA, h[s], du * Bsh[t][s]);
      y = fmaf(h[s], Csh[t][s], y);
    }
    float zv = xz[(rowbase + t) * (2 * D_INNERc) + D_INNERc + c];
    YFb[(rowbase + t) * D_INNERc + c] = (__bf16)((y + xv * dcoef) * silu_f(zv));
  }
}

// ---------------------------------------------------------------------------
// Launcher
// ---------------------------------------------------------------------------
extern "C" void kernel_launch(void* const* d_in, const int* in_sizes, int n_in,
                              void* d_out, int out_size, void* d_ws, size_t ws_size,
                              hipStream_t stream) {
  (void)in_sizes; (void)n_in; (void)out_size; (void)ws_size;
  const int M = MROWSc;
  const float* x = (const float*)d_in[0];
  float* out = (float*)d_out;
  float* ws = (float*)d_ws;

  // f32 workspace
  size_t oXZ = 0;
  size_t oXC = oXZ + (size_t)M * 2 * D_INNERc;
  size_t oXD = oXC + (size_t)M * D_INNERc;
  size_t oDT = oXD + (size_t)M * 64;
  size_t oP  = oDT + (size_t)M * D_INNERc;
  size_t oS  = oP  + (size_t)BATCHc * NCHUNKc * D_INNERc * D_STATEc;
  size_t oHI = oS  + (size_t)BATCHc * NCHUNKc * D_INNERc * D_STATEc;
  size_t f32end = oHI + (size_t)BATCHc * NCHUNKc * D_INNERc * D_STATEc;
  float *XZ = ws + oXZ, *XC = ws + oXC, *XD = ws + oXD, *DT = ws + oDT;
  float *Pb = ws + oP, *Sb = ws + oS, *HIb = ws + oHI;

  // bf16 workspace
  __bf16* wsb = (__bf16*)(ws + f32end);
  size_t bXF   = 0;
  size_t bWIN  = bXF   + (size_t)M * D_MODELc;
  size_t bWX   = bWIN  + (size_t)2048 * D_MODELc;
  size_t bWDT  = bWX   + (size_t)64 * D_INNERc;
  size_t bWOUT = bWDT  + (size_t)D_INNERc * DT_RANKc;
  size_t bXC   = bWOUT + (size_t)D_MODELc * D_INNERc;
  size_t bXD   = bXC   + (size_t)M * D_INNERc;
  size_t bYF   = bXD   + (size_t)M * 64;
  __bf16 *Xb = wsb + bXF, *WINb = wsb + bWIN, *WXb = wsb + bWX, *WDTb = wsb + bWDT;
  __bf16 *WOUTb = wsb + bWOUT, *XCb = wsb + bXC, *XDb = wsb + bXD, *YFb = wsb + bYF;

  // pack x once (shared by both directions; flip handled by row remap)
  pack_bf16_kernel<<<(M * D_MODELc / 4 + 255) / 256, 256, 0, stream>>>(x, Xb, M * D_MODELc / 4);

  for (int dir = 0; dir < 2; ++dir) {
    int base = 1 + dir * 9;
    const float* W_in   = (const float*)d_in[base + 0];
    const float* conv_w = (const float*)d_in[base + 1];
    const float* conv_b = (const float*)d_in[base + 2];
    const float* W_x    = (const float*)d_in[base + 3];
    const float* W_dt   = (const float*)d_in[base + 4];
    const float* b_dt   = (const float*)d_in[base + 5];
    const float* A_log  = (const float*)d_in[base + 6];
    const float* Dp     = (const float*)d_in[base + 7];
    const float* W_out  = (const float*)d_in[base + 8];

    // pack this direction's weights to bf16
    pack_bf16_kernel<<<(2048 * D_MODELc / 4 + 255) / 256, 256, 0, stream>>>(W_in, WINb, 2048 * D_MODELc / 4);
    pack_bf16_kernel<<<(64 * D_INNERc / 4 + 255) / 256, 256, 0, stream>>>(W_x, WXb, 64 * D_INNERc / 4);
    pack_bf16_kernel<<<(D_INNERc * DT_RANKc / 4 + 255) / 256, 256, 0, stream>>>(W_dt, WDTb, D_INNERc * DT_RANKc / 4);
    pack_bf16_kernel<<<(D_MODELc * D_INNERc / 4 + 255) / 256, 256, 0, stream>>>(W_out, WOUTb, D_MODELc * D_INNERc / 4);

    // GEMM1: xz = x(dir) @ W_in^T   [M,2048], K=512
    if (dir == 0)
      gemm_nt_bf16<2, 2, 0, false><<<dim3(2048 / 128, M / 128), 128, 0, stream>>>(
          Xb, D_MODELc, WINb, D_MODELc, XZ, 2 * D_INNERc, D_MODELc, 2048, nullptr, nullptr);
    else
      gemm_nt_bf16<2, 2, 0, true><<<dim3(2048 / 128, M / 128), 128, 0, stream>>>(
          Xb, D_MODELc, WINb, D_MODELc, XZ, 2 * D_INNERc, D_MODELc, 2048, nullptr, nullptr);

    // conv + SiLU: xc [M,1024] (f32 + bf16)
    conv_silu_kernel<<<(M * D_INNERc) / 256, 256, 0, stream>>>(XZ, conv_w, conv_b, XC, XCb);

    // GEMM2: x_dbl = xc @ W_x^T  [M,64], K=1024; dual-store f32 + bf16
    gemm_nt_bf16<4, 1, 4, false><<<dim3(1, M / 256), 128, 0, stream>>>(
        XCb, D_INNERc, WXb, D_INNERc, XD, 64, D_INNERc, 64, nullptr, XDb);

    // GEMM3: dt = softplus(x_dbl[:, :32] @ W_dt^T + b_dt)  [M,1024], K=32
    gemm_nt_bf16<2, 2, 1, false><<<dim3(D_INNERc / 128, M / 128), 128, 0, stream>>>(
        XDb, 64, WDTb, DT_RANKc, DT, D_INNERc, DT_RANKc, D_INNERc, b_dt, nullptr);

    // chunked selective scan
    scan_reduce_kernel<<<dim3(D_INNERc / 256, NCHUNKc, BATCHc), 256, 0, stream>>>(
        DT, XC, XD, A_log, Pb, Sb);
    scan_combine_kernel<<<(BATCHc * D_INNERc * D_STATEc) / 256, 256, 0, stream>>>(
        Pb, Sb, HIb);
    scan_apply_kernel<<<dim3(D_INNERc / 256, NCHUNKc, BATCHc), 256, 0, stream>>>(
        DT, XC, XD, A_log, HIb, XZ, Dp, YFb);

    // GEMM4: out = YF @ W_out^T  [M,512], K=1024 (fwd store / bwd flip-add)
    if (dir == 0)
      gemm_nt_bf16<2, 2, 2, false><<<dim3(D_MODELc / 128, M / 128), 128, 0, stream>>>(
          YFb, D_INNERc, WOUTb, D_INNERc, out, D_MODELc, D_INNERc, D_MODELc, nullptr, nullptr);
    else
      gemm_nt_bf16<2, 2, 3, false><<<dim3(D_MODELc / 128, M / 128), 128, 0, stream>>>(
          YFb, D_INNERc, WOUTb, D_INNERc, out, D_MODELc, D_INNERc, D_MODELc, nullptr, nullptr);
  }
}
